// ContrastiveLoss_747324309567
// MI455X (gfx1250) — matrix operand order repaired and checked by reading, not compiled, not run on hardware
//
#include <hip/hip_runtime.h>
#include <hip/hip_bf16.h>

typedef __attribute__((ext_vector_type(16))) _Float16 v16h;
typedef __attribute__((ext_vector_type(8)))  float    v8f;
typedef __attribute__((ext_vector_type(4)))  unsigned int v4u;
typedef __attribute__((ext_vector_type(8)))  int      v8i;
typedef __attribute__((ext_vector_type(4)))  int      v4i;

static constexpr int   NB      = 4096;       // batch
static constexpr int   ND      = 1024;       // feature dim
static constexpr float EPS     = 1e-6f;
static constexpr float INV_TAU = 1.0f / 0.07f;
static constexpr float EXTRA   = (float)NB * EPS + EPS;

#if defined(__HIP_DEVICE_COMPILE__) && __has_builtin(__builtin_amdgcn_tensor_load_to_lds)
#define USE_TDM 1
#else
#define USE_TDM 0
#endif

#if defined(__HIP_DEVICE_COMPILE__) && __has_builtin(__builtin_amdgcn_s_wait_tensorcnt)
#define WAIT_TENSORCNT(n) __builtin_amdgcn_s_wait_tensorcnt((short)(n))
#else
#define WAIT_TENSORCNT(n) asm volatile("s_wait_tensorcnt %0" ::"n"(n))
#endif

// ---------------------------------------------------------------------------
// Kernel 0: zero the reduction buffers (rowsum, colsum, diag) each launch.
// ---------------------------------------------------------------------------
__global__ void zero_kernel(float* __restrict__ p, int n) {
    int i = blockIdx.x * blockDim.x + threadIdx.x;
    if (i < n) p[i] = 0.0f;
}

// ---------------------------------------------------------------------------
// Kernel 1: row L2-normalize x and y, emit f16 rows into workspace.
// One block per row (blocks [0,NB) -> x, [NB,2NB) -> y). 256 threads/row.
// ---------------------------------------------------------------------------
__global__ __launch_bounds__(256)
void norm_kernel(const float* __restrict__ x, const float* __restrict__ y,
                 _Float16* __restrict__ xh, _Float16* __restrict__ yh) {
    const int  bid  = blockIdx.x;
    const bool isX  = bid < NB;
    const int  row  = isX ? bid : bid - NB;
    const float* src = isX ? x : y;
    _Float16*    dst = isX ? xh : yh;
    const int tid = threadIdx.x;

    float v[4];
    float ss = 0.0f;
#pragma unroll
    for (int i = 0; i < 4; ++i) {
        v[i] = src[(size_t)row * ND + tid + i * 256];
        ss += v[i] * v[i];
    }
#pragma unroll
    for (int o = 16; o > 0; o >>= 1) ss += __shfl_down(ss, o, 32);

    __shared__ float red[8];
    if ((tid & 31) == 0) red[tid >> 5] = ss;
    __syncthreads();
    if (tid == 0) {
        float t = 0.0f;
#pragma unroll
        for (int i = 0; i < 8; ++i) t += red[i];
        red[0] = 1.0f / fmaxf(sqrtf(t), EPS);
    }
    __syncthreads();
    const float scale = red[0];
#pragma unroll
    for (int i = 0; i < 4; ++i)
        dst[(size_t)row * ND + tid + i * 256] = (_Float16)(v[i] * scale);
}

// ---------------------------------------------------------------------------
// WMMA fragment loaders from LDS tiles laid out [rows][32] halfs.
// Layouts per CDNA5 ISA 7.12.2 (16-bit A 16x32, B 32x16), wave32.
// ---------------------------------------------------------------------------
struct FragU { union { v16h h; unsigned int u[8]; }; };

__device__ __forceinline__ v16h load_frag_a(const _Float16* s, int rowBase, int lane) {
    const int m     = rowBase + (lane & 15);
    const int khalf = (lane & 16) ? 8 : 0;           // lanes 16-31 take K offset +8
    const _Float16* p = s + m * 32;
    FragU f;
#pragma unroll
    for (int j = 0; j < 8; ++j) {
        const int k = ((j & 4) << 2) + khalf + ((j & 3) << 1);
        f.u[j] = *(const unsigned int*)(p + k);
    }
    return f.h;
}

__device__ __forceinline__ v16h load_frag_b(const _Float16* s, int colBase, int lane) {
    const int n     = colBase + (lane & 15);
    const int kbase = (lane & 16) ? 8 : 0;
    const _Float16* p = s + n * 32;
    FragU f;
#pragma unroll
    for (int j = 0; j < 8; ++j)
        f.u[j] = *(const unsigned int*)(p + ((j & 4) << 2) + kbase + ((j & 3) << 1));
    return f.h;
}

#if USE_TDM
// ---------------------------------------------------------------------------
// TDM: issue a 2D tile DMA (tile_dim0=32 halfs fastest, tile_dim1=128 rows)
// from a [NB x ND] f16 tensor (row stride ND) into LDS at ldsOff.
// D# per CDNA5 ISA ch.8: group0 {count=1, lds_addr, global_addr, type=2},
// group1 {data_size=2B, tensor dims/strides, tile dims}; groups 2/3 zero (2D).
// ---------------------------------------------------------------------------
__device__ __forceinline__ void tdm_load_tile(const _Float16* gsrc, unsigned ldsOff) {
    const unsigned long long ga = (unsigned long long)(size_t)gsrc;
    v4u g0;
    g0.x = 1u;                                            // count=1, user mode
    g0.y = ldsOff;                                        // lds_addr (bytes)
    g0.z = (unsigned)ga;                                  // global_addr[31:0]
    g0.w = (unsigned)((ga >> 32) & 0x01FFFFFFu) | 0x80000000u;  // addr[56:32] | type=2
    v8i g1;
    g1[0] = (int)(1u << 16);                              // data_size=1 (2 bytes)
    g1[1] = (int)(((unsigned)ND & 0xFFFFu) << 16);        // tensor_dim0 low16
    g1[2] = (int)(((unsigned)ND >> 16) | (((unsigned)NB & 0xFFFFu) << 16)); // dim0 hi | dim1 lo
    g1[3] = (int)(((unsigned)NB >> 16) | (32u << 16));    // dim1 hi | tile_dim0=32
    g1[4] = 128;                                          // tile_dim1=128, tile_dim2=0
    g1[5] = ND;                                           // tensor_dim0_stride[31:0]
    g1[6] = 0;                                            // stride0 hi | dim1_stride lo
    g1[7] = 0;
    v4i z4 = {0, 0, 0, 0};
#if __clang_major__ >= 23
    v8i z8 = {0, 0, 0, 0, 0, 0, 0, 0};
    __builtin_amdgcn_tensor_load_to_lds(g0, g1, z4, z4, z8, 0);
#else
    __builtin_amdgcn_tensor_load_to_lds(g0, g1, z4, z4, 0);
#endif
}
#endif

// ---------------------------------------------------------------------------
// Kernel 2: 128x128 tile of S = xn @ yn^T via v_wmma_f32_16x16x32_f16.
// TDM double-buffered global->LDS staging, fused exp(S/tau) epilogue,
// row/col partial sums via LDS atomics, diag grab.
// 256 threads = 8 waves arranged 2(M) x 4(N); wave tile 64x32 = 4x2 WMMA.
// ---------------------------------------------------------------------------
__global__ __launch_bounds__(256)
void gemm_exp_kernel(const _Float16* __restrict__ xh, const _Float16* __restrict__ yh,
                     float* __restrict__ rowsum, float* __restrict__ colsum,
                     float* __restrict__ diag) {
    __shared__ __align__(16) _Float16 As[2][128 * 32];
    __shared__ __align__(16) _Float16 Bs[2][128 * 32];
    __shared__ float rowAcc[128];
    __shared__ float colAcc[128];

    const int tid   = threadIdx.x;
    const int lane  = tid & 31;
    const int wave  = tid >> 5;
    const int waveM = wave >> 2;   // 0..1 -> 64-row slab
    const int waveN = wave & 3;    // 0..3 -> 32-col slab
    const int row0  = blockIdx.y * 128;
    const int col0  = blockIdx.x * 128;
    constexpr int NT = ND / 32;    // K-steps

    if (tid < 128) { rowAcc[tid] = 0.0f; colAcc[tid] = 0.0f; }

    v8f c[4][2] = {};

#if USE_TDM
    const unsigned asOff[2] = { (unsigned)(size_t)&As[0][0], (unsigned)(size_t)&As[1][0] };
    const unsigned bsOff[2] = { (unsigned)(size_t)&Bs[0][0], (unsigned)(size_t)&Bs[1][0] };

    if (wave == 0) {               // one wave drives the TDM (EXEC-independent DMA)
        tdm_load_tile(xh + (size_t)row0 * ND, asOff[0]);
        tdm_load_tile(yh + (size_t)col0 * ND, bsOff[0]);
    }
    for (int i = 0; i < NT; ++i) {
        const int cur = i & 1;
        if (wave == 0) {
            if (i + 1 < NT) {      // prefetch next tile into the other buffer
                tdm_load_tile(xh + (size_t)row0 * ND + (i + 1) * 32, asOff[cur ^ 1]);
                tdm_load_tile(yh + (size_t)col0 * ND + (i + 1) * 32, bsOff[cur ^ 1]);
                WAIT_TENSORCNT(2); // oldest pair (tile i) complete; TDM is in-order
            } else {
                WAIT_TENSORCNT(0);
            }
        }
        __syncthreads();           // tile `cur` visible to all waves

        v16h a[4], b[2];
#pragma unroll
        for (int mi = 0; mi < 4; ++mi) a[mi] = load_frag_a(As[cur], waveM * 64 + mi * 16, lane);
#pragma unroll
        for (int ni = 0; ni < 2; ++ni) b[ni] = load_frag_b(Bs[cur], waveN * 32 + ni * 16, lane);

#pragma unroll
        for (int mi = 0; mi < 4; ++mi)
#pragma unroll
            for (int ni = 0; ni < 2; ++ni)
                c[mi][ni] = __builtin_amdgcn_wmma_f32_16x16x32_f16(
                    false, a[mi], false, b[ni], (short)0, c[mi][ni], false, false);

        __syncthreads();           // all reads of `cur` done before it is re-DMAed
    }
#else
    for (int kb = 0; kb < ND; kb += 32) {
        __syncthreads();
#pragma unroll
        for (int p = 0; p < 2; ++p) {
            const int chunk = p * 256 + tid;
            const int row   = chunk >> 2;
            const int c4    = chunk & 3;
            const float4 va = *(const float4*)(xh + (size_t)(row0 + row) * ND + kb + c4 * 8);
            *(float4*)(As[0] + chunk * 8) = va;
            const float4 vb = *(const float4*)(yh + (size_t)(col0 + row) * ND + kb + c4 * 8);
            *(float4*)(Bs[0] + chunk * 8) = vb;
        }
        __syncthreads();

        v16h a[4], b[2];
#pragma unroll
        for (int mi = 0; mi < 4; ++mi) a[mi] = load_frag_a(As[0], waveM * 64 + mi * 16, lane);
#pragma unroll
        for (int ni = 0; ni < 2; ++ni) b[ni] = load_frag_b(Bs[0], waveN * 32 + ni * 16, lane);

#pragma unroll
        for (int mi = 0; mi < 4; ++mi)
#pragma unroll
            for (int ni = 0; ni < 2; ++ni)
                c[mi][ni] = __builtin_amdgcn_wmma_f32_16x16x32_f16(
                    false, a[mi], false, b[ni], (short)0, c[mi][ni], false, false);
    }
#endif

    // ---- fused epilogue: E = exp(S/tau); accumulate row/col sums; grab diag
    float colLocal[2] = {0.0f, 0.0f};
#pragma unroll
    for (int mi = 0; mi < 4; ++mi) {
#pragma unroll
        for (int r = 0; r < 8; ++r) {
            const int rowL = waveM * 64 + mi * 16 + r + ((lane & 16) ? 8 : 0);
            float es = 0.0f;
#pragma unroll
            for (int ni = 0; ni < 2; ++ni) {
                const float e = __expf(c[mi][ni][r] * INV_TAU);
                es += e;
                colLocal[ni] += e;
                const int gi = row0 + rowL;
                const int gj = col0 + waveN * 32 + ni * 16 + (lane & 15);
                if (gi == gj) diag[gi] = e;
            }
            atomicAdd(&rowAcc[rowL], es);
        }
    }
    const int nlane = lane & 15;
    atomicAdd(&colAcc[waveN * 32 + nlane],      colLocal[0]);
    atomicAdd(&colAcc[waveN * 32 + 16 + nlane], colLocal[1]);

    __syncthreads();
    if (tid < 128) {
        atomicAdd(&rowsum[row0 + tid], rowAcc[tid]);
        atomicAdd(&colsum[col0 + tid], colAcc[tid]);
    }
}

// ---------------------------------------------------------------------------
// Kernel 3: loss = -1/(2B) * sum_i [ 2 log diag_i - log(row_i+extra) - log(col_i+extra) ]
// ---------------------------------------------------------------------------
__global__ __launch_bounds__(256)
void final_kernel(const float* __restrict__ rowsum, const float* __restrict__ colsum,
                  const float* __restrict__ diag, float* __restrict__ out) {
    const int tid = threadIdx.x;
    float acc = 0.0f;
    for (int i = tid; i < NB; i += 256)
        acc += 2.0f * __logf(diag[i]) - __logf(rowsum[i] + EXTRA) - __logf(colsum[i] + EXTRA);

#pragma unroll
    for (int o = 16; o > 0; o >>= 1) acc += __shfl_down(acc, o, 32);
    __shared__ float red[8];
    if ((tid & 31) == 0) red[tid >> 5] = acc;
    __syncthreads();
    if (tid == 0) {
        float t = 0.0f;
#pragma unroll
        for (int i = 0; i < 8; ++i) t += red[i];
        out[0] = t * (-1.0f / (2.0f * (float)NB));
    }
}

// ---------------------------------------------------------------------------
extern "C" void kernel_launch(void* const* d_in, const int* in_sizes, int n_in,
                              void* d_out, int out_size, void* d_ws, size_t ws_size,
                              hipStream_t stream) {
    const float* x = (const float*)d_in[0];
    const float* y = (const float*)d_in[1];

    char* ws = (char*)d_ws;
    _Float16* xh     = (_Float16*)ws;                                   // 8 MB
    _Float16* yh     = (_Float16*)(ws + (size_t)NB * ND * 2);           // 8 MB
    float*    rowsum = (float*)(ws + (size_t)NB * ND * 4);              // 16 KB
    float*    colsum = rowsum + NB;                                     // 16 KB
    float*    diag   = colsum + NB;                                     // 16 KB

    zero_kernel<<<(3 * NB + 255) / 256, 256, 0, stream>>>(rowsum, 3 * NB);
    norm_kernel<<<2 * NB, 256, 0, stream>>>(x, y, xh, yh);
    gemm_exp_kernel<<<dim3(NB / 128, NB / 128), 256, 0, stream>>>(xh, yh, rowsum, colsum, diag);
    final_kernel<<<1, 256, 0, stream>>>(rowsum, colsum, diag, (float*)d_out);
}
